// GraphSelfAttention_33526514712911
// MI455X (gfx1250) — matrix operand order repaired and checked by reading, not compiled, not run on hardware
//
#include <hip/hip_runtime.h>

typedef __attribute__((ext_vector_type(4)))  unsigned int u32x4;
typedef __attribute__((ext_vector_type(4)))  float        f32x4;
typedef __attribute__((ext_vector_type(8)))  float        v8f;
typedef __attribute__((ext_vector_type(8)))  __bf16       bf16x8;
typedef __attribute__((ext_vector_type(16))) __bf16       v16bf;

union Frag { v16bf v; unsigned int u[8]; u32x4 q[2]; };

#define WMMA_BF16(a, b, c) \
  __builtin_amdgcn_wmma_f32_16x16x32_bf16(false, (a), false, (b), (short)0, (c), false, false)

__device__ __forceinline__ void async_b128x4(unsigned lds, unsigned long long ga) {
  asm volatile(
      "global_load_async_to_lds_b128 %0, %1, off\n\t"
      "global_load_async_to_lds_b128 %0, %1, off offset:16\n\t"
      "global_load_async_to_lds_b128 %0, %1, off offset:32\n\t"
      "global_load_async_to_lds_b128 %0, %1, off offset:48"
      :: "v"(lds), "v"(ga) : "memory");
}

// ---------------------------------------------------------------------------
// Generic GEMM: C[M,Nc] = scale * (A[M,K] @ B[K,Nc] + bias)
// fp32 in HBM -> bf16 LDS staging -> WMMA f32 accumulate.
// 64x64 tile / workgroup, BK=32, 8 waves (wave32), 2 accum tiles per wave.
// ---------------------------------------------------------------------------
__global__ __launch_bounds__(256) void gemm_wmma(
    const float* __restrict__ A, const float* __restrict__ B,
    const float* __restrict__ bias,
    float* __restrict__ outF, __bf16* __restrict__ outB,
    __bf16* __restrict__ outT, int ldT,
    int M, int Nc, int K, float scale)
{
  __shared__ __bf16 As[64][32];   // [m][k]
  __shared__ __bf16 Bs[64][32];   // [n][k] (transposed: B-frag pairs contiguous)
  const int tid  = threadIdx.x;
  const int w    = tid >> 5, lane = tid & 31, ln = lane & 15, hi = lane >> 4;
  const int mBase = blockIdx.x * 64, nBase = blockIdx.y * 64;
  const int rA = (w & 3) * 16;
  const int nT = (w >> 2) * 32;

  const int ar = tid >> 2, ac = (tid & 3) * 8;
  const int kkB = tid >> 3, n0B = (tid & 7) * 8;

  v8f acc0 = {}, acc1 = {};
  const int KT = (K + 31) / 32;
  for (int kt = 0; kt < KT; ++kt) {
    const int k0g = kt * 32;
    const bool full = (k0g + 32 <= K);
    {   // stage A 64x32
      const float* Ar = A + (size_t)(mBase + ar) * K + k0g;
      if (full) {
        f32x4 a0 = *(const f32x4*)(Ar + ac);
        f32x4 a1 = *(const f32x4*)(Ar + ac + 4);
        bf16x8 t;
        #pragma unroll
        for (int j = 0; j < 4; ++j) { t[j] = (__bf16)a0[j]; t[4 + j] = (__bf16)a1[j]; }
        *(bf16x8*)&As[ar][ac] = t;
      } else {
        bf16x8 t;
        #pragma unroll
        for (int j = 0; j < 8; ++j) {
          int gk = k0g + ac + j;
          t[j] = (__bf16)(gk < K ? Ar[ac + j] : 0.0f);
        }
        *(bf16x8*)&As[ar][ac] = t;
      }
    }
    {   // stage B 32x64 -> Bs[n][k]
      const int gk = k0g + kkB;
      if (gk < K) {
        const float* Br = B + (size_t)gk * Nc + nBase + n0B;
        f32x4 b0 = *(const f32x4*)Br;
        f32x4 b1 = *(const f32x4*)(Br + 4);
        #pragma unroll
        for (int j = 0; j < 4; ++j) {
          Bs[n0B + j][kkB]     = (__bf16)b0[j];
          Bs[n0B + 4 + j][kkB] = (__bf16)b1[j];
        }
      } else {
        #pragma unroll
        for (int j = 0; j < 8; ++j) Bs[n0B + j][kkB] = (__bf16)0.0f;
      }
    }
    __syncthreads();
    Frag af, bf0, bf1;
    #pragma unroll
    for (int p = 0; p < 8; ++p) {
      int ka = 16 * hi + 2 * p;
      af.u[p]  = *(const unsigned int*)&As[rA + ln][ka];
      int kb   = 16 * hi + 2 * p;
      bf0.u[p] = *(const unsigned int*)&Bs[nT + ln][kb];
      bf1.u[p] = *(const unsigned int*)&Bs[nT + 16 + ln][kb];
    }
    acc0 = WMMA_BF16(af.v, bf0.v, acc0);
    acc1 = WMMA_BF16(af.v, bf1.v, acc1);
    __syncthreads();
  }
  #pragma unroll
  for (int t = 0; t < 2; ++t) {
    v8f acc = t ? acc1 : acc0;
    const int col = nBase + nT + 16 * t + ln;
    const float bv = bias ? bias[col] : 0.0f;
    #pragma unroll
    for (int r = 0; r < 8; ++r) {
      const int row = mBase + rA + r + 8 * hi;
      const float val = (acc[r] + bv) * scale;
      if (outF) outF[(size_t)row * Nc + col] = val;
      if (outB) outB[(size_t)row * Nc + col] = (__bf16)val;
      if (outT) outT[(size_t)col * ldT + row] = (__bf16)val;
    }
  }
}

// ---------------------------------------------------------------------------
// Pack bool mask into bitmask words: bit i of word j = mask[j*32+i]
// ---------------------------------------------------------------------------
__global__ __launch_bounds__(256) void pack_mask(const unsigned char* __restrict__ m,
                                                 unsigned* __restrict__ mb) {
  const size_t idx = (size_t)blockIdx.x * blockDim.x + threadIdx.x;
  const unsigned word = __builtin_amdgcn_ballot_w32(m[idx] != 0);
  if ((threadIdx.x & 31) == 0) mb[idx >> 5] = word;
}

// ---------------------------------------------------------------------------
// Flash attention: workgroup = (head, 128-query block); 8 waves each own a
// 16-query tile. Workgroup-uniform loop over 256 key-blocks (32 keys) with
// triple-buffered async-to-LDS staging of K and V^T (1 barrier / iteration).
// ---------------------------------------------------------------------------
#define NKEYS 8192
__global__ __launch_bounds__(256) void attn_flash(
    const __bf16* __restrict__ Qall, const __bf16* __restrict__ Kall,
    const __bf16* __restrict__ Vt,   const unsigned* __restrict__ mbits,
    float* __restrict__ ctx)
{
  __shared__ __bf16 Kbuf[3][32][264];   // [key][d], padded rows (bank spread)
  __shared__ __bf16 Vbuf[3][256][40];   // [d][key], padded rows
  __shared__ __bf16 Ps[8][16 * 32];

  const int tid = threadIdx.x;
  const int w = tid >> 5, lane = tid & 31, ln = lane & 15, hi = lane >> 4;
  const int qBase = blockIdx.x * 128 + w * 16;
  const int head  = blockIdx.y;
  const __bf16* Q  = Qall + ((size_t)head * NKEYS + qBase) * 256;
  const __bf16* Km = Kall + (size_t)head * NKEYS * 256;
  const __bf16* Vh = Vt   + (size_t)head * 256 * NKEYS;

  const int krow = tid >> 3, kcol = (tid & 7) * 32;   // K staging coords

  auto stage = [&](int buf, int nb) {
    async_b128x4((unsigned)(size_t)&Kbuf[buf][krow][kcol],
                 (unsigned long long)(size_t)(Km + (size_t)(nb + krow) * 256 + kcol));
    async_b128x4((unsigned)(size_t)&Vbuf[buf][tid][0],
                 (unsigned long long)(size_t)(Vh + (size_t)tid * NKEYS + nb));
  };

  // Q as 8 A-fragments, read directly from global (two 16B runs per lane/chunk)
  Frag qf[8];
  #pragma unroll
  for (int c = 0; c < 8; ++c) {
    const __bf16* qp = Q + ln * 256 + c * 32 + 16 * hi;
    qf[c].q[0] = *(const u32x4*)qp;
    qf[c].q[1] = *(const u32x4*)(qp + 8);
  }

  float rm[8], rs[8];
  v8f acc[16] = {};
  #pragma unroll
  for (int r = 0; r < 8; ++r) { rm[r] = -1e38f; rs[r] = 0.0f; }

  stage(0, 0);                      // prologue: blocks 0 and 1 in flight
  stage(1, 32);

  for (int kb = 0; kb < NKEYS / 32; ++kb) {
    const int cur = kb % 3;
    const int keyBase = kb * 32;
    // wait for current block (next block's 8 ops may stay in flight), then sync
    if (kb < NKEYS / 32 - 1) asm volatile("s_wait_asynccnt 0x8" ::: "memory");
    else                     asm volatile("s_wait_asynccnt 0x0" ::: "memory");
    __syncthreads();
    // prefetch block kb+2 into the buffer everyone just finished reading
    if (kb + 2 < NKEYS / 32) stage((kb + 2) % 3, keyBase + 64);

    v8f s0 = {}, s1 = {};
    #pragma unroll
    for (int c = 0; c < 8; ++c) {        // S = Q @ K^T  (two 16x16 tiles)
      Frag k0f, k1f;
      const __bf16* p0 = &Kbuf[cur][ln][c * 32 + 16 * hi];
      const __bf16* p1 = &Kbuf[cur][16 + ln][c * 32 + 16 * hi];
      k0f.q[0] = *(const u32x4*)p0; k0f.q[1] = *(const u32x4*)(p0 + 8);
      k1f.q[0] = *(const u32x4*)p1; k1f.q[1] = *(const u32x4*)(p1 + 8);
      s0 = WMMA_BF16(qf[c].v, k0f.v, s0);
      s1 = WMMA_BF16(qf[c].v, k1f.v, s1);
    }
    #pragma unroll
    for (int r = 0; r < 8; ++r) {        // masked online softmax per query row
      const int rowq = qBase + r + 8 * hi;
      const unsigned mw = mbits[(size_t)rowq * (NKEYS / 32) + kb];
      float sv0 = ((mw >> ln) & 1u)        ? s0[r] : -1e38f;
      float sv1 = ((mw >> (16 + ln)) & 1u) ? s1[r] : -1e38f;
      float bm = fmaxf(sv0, sv1);
      #pragma unroll
      for (int off = 1; off < 16; off <<= 1) bm = fmaxf(bm, __shfl_xor(bm, off));
      float nm = fmaxf(rm[r], bm);
      float al = __expf(rm[r] - nm);
      float e0 = __expf(sv0 - nm), e1 = __expf(sv1 - nm);
      float ps = e0 + e1;
      #pragma unroll
      for (int off = 1; off < 16; off <<= 1) ps += __shfl_xor(ps, off);
      rs[r] = rs[r] * al + ps;
      rm[r] = nm;
      if (__builtin_amdgcn_ballot_w32(al != 1.0f)) {   // skip no-op rescale
        #pragma unroll
        for (int t = 0; t < 16; ++t) acc[t][r] *= al;
      }
      Ps[w][(r + 8 * hi) * 32 + ln]      = (__bf16)e0;   // C-layout -> LDS
      Ps[w][(r + 8 * hi) * 32 + 16 + ln] = (__bf16)e1;
    }
    Frag pf;                              // reload P as A-fragment (transpose)
    #pragma unroll
    for (int p = 0; p < 8; ++p) {
      int k0 = 16 * hi + 2 * p;
      pf.u[p] = *(const unsigned int*)&Ps[w][ln * 32 + k0];
    }
    #pragma unroll
    for (int t = 0; t < 16; ++t) {        // ctx += P @ V
      Frag vf;
      const __bf16* pv = &Vbuf[cur][t * 16 + ln][16 * hi];
      vf.q[0] = *(const u32x4*)pv; vf.q[1] = *(const u32x4*)(pv + 8);
      acc[t] = WMMA_BF16(pf.v, vf.v, acc[t]);
    }
  }

  // epilogue: normalize and store (wave owns complete rows -> no merge)
  float* co = ctx + ((size_t)head * NKEYS + qBase) * 256;
  #pragma unroll
  for (int t = 0; t < 16; ++t)
    #pragma unroll
    for (int r = 0; r < 8; ++r)
      co[(size_t)(r + 8 * hi) * 256 + t * 16 + ln] = acc[t][r] / rs[r];
}

__global__ void mean_heads(const float* __restrict__ ctx, float* __restrict__ outm) {
  const size_t i = (size_t)blockIdx.x * blockDim.x + threadIdx.x;
  const size_t T = (size_t)NKEYS * 256;
  outm[i] = 0.25f * (ctx[i] + ctx[i + T] + ctx[i + 2 * T] + ctx[i + 3 * T]);
}

__global__ __launch_bounds__(256) void ln_resid(
    const float* __restrict__ hs, const float* __restrict__ h,
    const float* __restrict__ lnw, const float* __restrict__ lnb,
    float* __restrict__ out)
{
  const int row = blockIdx.x * 8 + (threadIdx.x >> 5);
  const int lane = threadIdx.x & 31;
  const float* a = hs + (size_t)row * 256;
  const float* b = h  + (size_t)row * 256;
  float v[8], sum = 0.0f, sq = 0.0f;
  #pragma unroll
  for (int i = 0; i < 8; ++i) {
    const int c = lane + 32 * i;
    const float y = a[c] + b[c];
    v[i] = y; sum += y; sq += y * y;
  }
  #pragma unroll
  for (int off = 1; off < 32; off <<= 1) { sum += __shfl_xor(sum, off); sq += __shfl_xor(sq, off); }
  const float u = sum * (1.0f / 256.0f);
  const float s = sq * (1.0f / 256.0f) - u * u;
  const float inv = rsqrtf(s + 1e-12f);
  float* o = out + (size_t)row * 256;
  #pragma unroll
  for (int i = 0; i < 8; ++i) {
    const int c = lane + 32 * i;
    o[c] = lnw[c] * (v[i] - u) * inv + lnb[c];
  }
}

extern "C" void kernel_launch(void* const* d_in, const int* in_sizes, int n_in,
                              void* d_out, int out_size, void* d_ws, size_t ws_size,
                              hipStream_t stream) {
  (void)in_sizes; (void)n_in; (void)out_size; (void)ws_size;
  const float* x    = (const float*)d_in[0];
  const unsigned char* mask = (const unsigned char*)d_in[1];  // np.bool_ layout
  const float* AE_w = (const float*)d_in[2];
  const float* AE_b = (const float*)d_in[3];
  const float* Wq   = (const float*)d_in[4];
  const float* bq   = (const float*)d_in[5];
  const float* Wk   = (const float*)d_in[6];
  const float* bk   = (const float*)d_in[7];
  const float* Wv   = (const float*)d_in[8];
  const float* bv   = (const float*)d_in[9];
  const float* dw   = (const float*)d_in[10];
  const float* db   = (const float*)d_in[11];
  const float* lnw  = (const float*)d_in[12];
  const float* lnb  = (const float*)d_in[13];

  const int N = 8192, D = 256, IN = 2000, H = 4;
  char* ws = (char*)d_ws;
  size_t off = 0;
  auto alloc = [&](size_t bytes) -> void* {
    void* p = ws + off; off += (bytes + 255) & ~(size_t)255; return p;
  };
  float*    hbuf = (float*)   alloc((size_t)N * D * 4);     // h = x@AE_w + b
  float*    hsb  = (float*)   alloc((size_t)N * D * 4);     // dense output
  float*    outm = (float*)   alloc((size_t)N * D * 4);     // head-mean ctx
  float*    ctx  = (float*)   alloc((size_t)H * N * D * 4); // per-head ctx
  __bf16*   qb   = (__bf16*)  alloc((size_t)H * N * D * 2); // Q * (1/sqrt(D))
  __bf16*   kbf  = (__bf16*)  alloc((size_t)H * N * D * 2); // K
  __bf16*   vt   = (__bf16*)  alloc((size_t)H * N * D * 2); // V^T  [D][N]
  unsigned* mbw  = (unsigned*)alloc((size_t)N * (N / 32) * 4); // packed mask

  dim3 gg(N / 64, D / 64), bb(256);
  pack_mask<<<dim3(((size_t)N * N) / 256), bb, 0, stream>>>(mask, mbw);
  gemm_wmma<<<gg, bb, 0, stream>>>(x, AE_w, AE_b, hbuf, nullptr, nullptr, 0,
                                   N, D, IN, 1.0f);
  for (int hh = 0; hh < H; ++hh) {
    gemm_wmma<<<gg, bb, 0, stream>>>(hbuf, Wq + (size_t)hh * D * D, bq + hh * D,
        nullptr, qb + (size_t)hh * N * D, nullptr, 0, N, D, D, 0.0625f);
    gemm_wmma<<<gg, bb, 0, stream>>>(hbuf, Wk + (size_t)hh * D * D, bk + hh * D,
        nullptr, kbf + (size_t)hh * N * D, nullptr, 0, N, D, D, 1.0f);
    gemm_wmma<<<gg, bb, 0, stream>>>(hbuf, Wv + (size_t)hh * D * D, bv + hh * D,
        nullptr, nullptr, vt + (size_t)hh * D * N, N, N, D, D, 1.0f);
  }
  attn_flash<<<dim3(N / 128, H), bb, 0, stream>>>(qb, kbf, vt, mbw, ctx);
  mean_heads<<<dim3((N * D) / 256), bb, 0, stream>>>(ctx, outm);
  gemm_wmma<<<gg, bb, 0, stream>>>(outm, dw, db, hsb, nullptr, nullptr, 0,
                                   N, D, D, 1.0f);
  ln_resid<<<dim3(N / 8), bb, 0, stream>>>(hsb, hbuf, lnw, lnb, (float*)d_out);
}
